// CFRMDecoder_56229711839236
// MI455X (gfx1250) — matrix operand order repaired and checked by reading, not compiled, not run on hardware
//
#include <hip/hip_runtime.h>
#include <hip/hip_bf16.h>
#include <math.h>

// ---------------- problem constants (from reference) ----------------
#define Vv 32000
#define Cc 32
#define Hh 256
#define Ee 256
#define Bb 8
#define Ss 128
#define Mm (Bb * Ss)      // 1024 rows for the batched GEMMs
#define G3H (3 * Hh)      // 768
#define CH (Cc * Hh)      // 8192
#define FEAT (3 * Hh + 4) // 772
#define CIN (2 * Hh + 4)  // 516
#define Kk 256            // shared K of both big GEMMs (== Hh == Ee)
#define EPSf 1e-4f

typedef __attribute__((ext_vector_type(16))) __bf16        v16bf;
typedef __attribute__((ext_vector_type(8)))  float         v8f;
typedef __attribute__((ext_vector_type(4)))  unsigned int  v4u;
typedef __attribute__((ext_vector_type(8)))  int           v8i;
typedef __attribute__((ext_vector_type(4)))  int           v4i;

// ---------------------------------------------------------------
// Kernel 1: embedding gather, emitted directly as bf16 GEMM operand
// ---------------------------------------------------------------
__global__ void embed_bf16_kernel(const int* __restrict__ tokens,
                                  const float* __restrict__ emb,
                                  __bf16* __restrict__ Xbf) {
    const int m = blockIdx.x;
    const int t = threadIdx.x;
    Xbf[(size_t)m * Ee + t] = (__bf16)emb[(size_t)tokens[m] * Ee + t];
}

// fp32 -> bf16 straight convert (src already N x K row-major)
__global__ void convert_bf16_kernel(const float* __restrict__ src,
                                    __bf16* __restrict__ dst) {
    const size_t i = (size_t)blockIdx.x * 256 + threadIdx.x;
    dst[i] = (__bf16)src[i];
}

// fp32 K x N -> bf16 N x K (transpose-convert), once per call.
__global__ void transpose_bf16_kernel(const float* __restrict__ src,
                                      __bf16* __restrict__ dst,
                                      int N, int K) {
    const int n = blockIdx.x;
    const int k = threadIdx.x;
    dst[(size_t)n * K + k] = (__bf16)src[(size_t)k * N + n];
}

// ---------------------------------------------------------------
// WMMA bf16 GEMM:  C[m,n] = sum_k A[m,k]*B[n,k] + bias[n]
//   A: M x K bf16 row-major,  B: N x K bf16 row-major (pre-transposed)
// Block tile: 16 x 512.  The 16x256 bf16 A panel (8 KB) is staged in
// LDS once per block via the Tensor Data Mover (tensor_load_to_lds +
// s_wait_tensorcnt), then shared by all 8 waves.  Each wave computes a
// 16x64 strip: one LDS A fragment feeds 4 WMMAs per K-step, K fully
// unrolled -> 32 static v_wmma_f32_16x16x32_bf16 per wave.
// ---------------------------------------------------------------
__global__ __launch_bounds__(256)
void wmma_gemm_bf16_kernel(const __bf16* __restrict__ Abf,
                           const __bf16* __restrict__ Bbf,
                           const float* __restrict__ bias,
                           float* __restrict__ Cout,
                           int M, int N, int nblkn) {
    __shared__ __bf16 s_a[16 * Kk];            // 8 KB A panel
    const int wave = threadIdx.x >> 5;
    const int lane = threadIdx.x & 31;
    const int bm = blockIdx.x / nblkn;
    const int bn = blockIdx.x % nblkn;
    const int tm = bm << 4;

    // ---- stage A panel (16 rows x 256 K) into LDS via TDM ----
#if __has_builtin(__builtin_amdgcn_tensor_load_to_lds)
    if (threadIdx.x < 32) {                    // wave 0 issues the DMA
        const unsigned long long ga =
            (unsigned long long)(const void*)(Abf + (size_t)tm * Kk);
        const unsigned ldsoff = (unsigned)(size_t)(void*)s_a; // LDS byte offset
        // D# group0: count=1 | lds_addr | global_addr[56:0] | type=2
        v4u g0 = { 1u, ldsoff, (unsigned)ga,
                   (unsigned)((ga >> 32) & 0x01FFFFFFull) | 0x80000000u };
        // D# group1: data_size=1(2B); tensor_dim0=256; tensor_dim1=16;
        //            tile_dim0=256; tile_dim1=16; dim0_stride=256
        v8i g1 = { 0x00010000,        // wg_mask=0 | data_size=1<<16
                   0x01000000,        // tensor_dim0[15:0]=256 @ [63:48]
                   0x00100000,        // tensor_dim1[15:0]=16  @ [95:80]
                   0x01000000,        // tile_dim0=256         @ [127:112]
                   16,                // tile_dim1=16, tile_dim2=0
                   256,               // tensor_dim0_stride lo32
                   0, 0 };
        v4i g2 = { 0, 0, 0, 0 };      // 2D tensor: groups 2/3 unused
        v4i g3 = { 0, 0, 0, 0 };
        v8i gx = { 0, 0, 0, 0, 0, 0, 0, 0 };   // extended group (unused, 2D)
        __builtin_amdgcn_tensor_load_to_lds(g0, g1, g2, g3, gx, 0);
        __builtin_amdgcn_s_wait_tensorcnt(0);
    }
#else
    for (int i = threadIdx.x; i < 16 * Kk; i += 256)
        s_a[i] = Abf[(size_t)tm * Kk + i];
#endif
    __syncthreads();

    const int tn = bn * 512 + wave * 64;       // wave-uniform strip base
    if (tn < N) {                              // full-wave predicate
        const int row = lane & 15;             // A row / B,C column
        const int hi  = lane >> 4;             // lane-group K select

        v8f acc[4] = {v8f{}, v8f{}, v8f{}, v8f{}};
        const __bf16* arow = s_a + row * Kk;   // LDS-resident A row

#pragma unroll
        for (int k0 = 0; k0 < Kk; k0 += 32) {
            // A fragment (16x32) from LDS: two 16B ds loads per lane.
            v16bf a;
            {
                const __bf16* ap = arow + k0 + hi * 16;
#pragma unroll
                for (int j = 0; j < 8; ++j) { a[j] = ap[j]; a[j + 8] = ap[8 + j]; }
            }
            const int kb = k0 + hi * 16;
#pragma unroll
            for (int u = 0; u < 4; ++u) {
                // B fragment (32x16): lane = column, 16 contiguous bf16.
                const __bf16* bp = Bbf + (size_t)(tn + u * 16 + row) * Kk + kb;
                if (k0 == 0)   // stream hint for the rest of this K-row
                    __builtin_prefetch(bp + 128, 0, 1);
                v16bf b;
#pragma unroll
                for (int j = 0; j < 16; ++j) b[j] = bp[j];
                acc[u] = __builtin_amdgcn_wmma_f32_16x16x32_bf16(
                    false, a, false, b, (short)0, acc[u], false, false);
            }
        }

        // D store: VGPR j -> row tm+j (lanes 0-15) / tm+j+8 (lanes 16-31)
#pragma unroll
        for (int u = 0; u < 4; ++u) {
            const int n = tn + u * 16 + row;
            const float bv = bias ? bias[n] : 0.0f;
#pragma unroll
            for (int j = 0; j < 8; ++j) {
                const int m = tm + j + hi * 8;
                Cout[(size_t)m * N + n] = acc[u][j] + bv;
            }
        }
    }
}

// ---------------------------------------------------------------
// Kernel 3: GRU recurrence.  One block per batch row, thread = hidden
// unit, h state in LDS.  GI already holds x@wih.T + bih for all t.
// ---------------------------------------------------------------
__global__ __launch_bounds__(256)
void gru_scan_kernel(const float* __restrict__ GI,
                     const float* __restrict__ whh,
                     const float* __restrict__ bhh,
                     float* __restrict__ HL) {
    const int b = blockIdx.x;
    const int u = threadIdx.x;
    __shared__ float h_sh[Hh];
    h_sh[u] = 0.0f;
    __syncthreads();

    const float* wr = whh + (size_t)u * Hh;
    const float* wz = whh + (size_t)(Hh + u) * Hh;
    const float* wn = whh + (size_t)(2 * Hh + u) * Hh;
    const float br = bhh[u], bz = bhh[Hh + u], bn = bhh[2 * Hh + u];

    for (int s = 0; s < Ss; ++s) {
        const float* gi = GI + ((size_t)b * Ss + s) * G3H;
        const float ir = gi[u], iz = gi[Hh + u], inx = gi[2 * Hh + u];
        float hr = br, hz = bz, hn = bn;
        for (int k = 0; k < Hh; ++k) {
            const float hv = h_sh[k];
            hr += wr[k] * hv; hz += wz[k] * hv; hn += wn[k] * hv;
        }
        const float hprev = h_sh[u];
        const float r = 1.0f / (1.0f + expf(-(ir + hr)));
        const float z = 1.0f / (1.0f + expf(-(iz + hz)));
        const float n = tanhf(inx + r * hn);
        const float hnew = (1.0f - z) * n + z * hprev;
        __syncthreads();
        h_sh[u] = hnew;
        __syncthreads();
        HL[((size_t)b * Ss + s) * Hh + u] = hnew;
    }
}

// ---------------------------------------------------------------
// helpers for the cluster scan
// ---------------------------------------------------------------
__device__ __forceinline__ float sigmoidf_(float x) { return 1.0f / (1.0f + expf(-x)); }
__device__ __forceinline__ float softplusf_(float x) { return (x > 20.0f) ? x : log1pf(expf(x)); }

// summarize(): fills alpha[C], core[H], scal[0]=unc scal[1]=div scal[2]=energy
// scal[3]=entropy, *idxp = argmax(alpha).  All-thread collective.
__device__ __forceinline__ void summarize_block(
    int t, float (*cent)[Hh], const float* spr, const float* mass,
    float* alpha, float* core, float* red, float* scal, int* idxp) {
    __syncthreads();
    if (t == 0) {
        float score[Cc];
        float mx = -1e30f;
        for (int c = 0; c < Cc; ++c) {
            const float prec = 1.0f / (spr[c] + EPSf);
            score[c] = mass[c] + logf(prec + EPSf);
            mx = fmaxf(mx, score[c]);
        }
        float se = 0.0f;
        for (int c = 0; c < Cc; ++c) { const float e = expf(score[c] - mx); alpha[c] = e; se += e; }
        float unc = 0.0f, ent = 0.0f, ba = -1.0f;
        int best = 0;
        for (int c = 0; c < Cc; ++c) {
            const float a = alpha[c] / se;
            alpha[c] = a;
            unc += a * spr[c];
            ent -= a * logf(fmaxf(a, 1e-8f));
            if (a > ba) { ba = a; best = c; }
        }
        scal[0] = unc; scal[3] = ent; *idxp = best;
        float mmx = -1e30f;
        for (int c = 0; c < Cc; ++c) mmx = fmaxf(mmx, mass[c]);
        float es = 0.0f;
        for (int c = 0; c < Cc; ++c) es += expf(mass[c] - mmx);
        scal[2] = mmx + logf(es);
    }
    __syncthreads();
    {   // core[h] = sum_c alpha[c]*centers[c,h]
        float cv = 0.0f;
        for (int c = 0; c < Cc; ++c) cv += alpha[c] * cent[c][t];
        core[t] = cv;
    }
    __syncthreads();
    {   // sq_dist: 8 threads per cluster, each sums 32 h values
        const int c = t >> 3, j = t & 7;
        float p = 0.0f;
        for (int h = j * 32; h < j * 32 + 32; ++h) {
            const float d = cent[c][h] - core[h];
            p += d * d;
        }
        red[t] = p;
    }
    __syncthreads();
    if (t == 0) {
        float dv = 0.0f;
        for (int c = 0; c < Cc; ++c) {
            float s = 0.0f;
            for (int j = 0; j < 8; ++j) s += red[c * 8 + j];
            dv += alpha[c] * (s / (float)Hh);
        }
        scal[1] = dv;
    }
    __syncthreads();
}

// ---------------------------------------------------------------
// Kernel 4: cluster-field recurrence.  One block per batch row; all
// state LDS-resident (centers 32KB << 320KB/WGP).  Produces bf16
// H1[b,s,:] = gelu(feat @ out_w1 + out_b1) for the WMMA projection.
// ---------------------------------------------------------------
__global__ __launch_bounds__(256)
void cluster_scan_kernel(const int* __restrict__ tokens,
                         const float* __restrict__ HL,
                         const float* __restrict__ ctrl_w1, const float* __restrict__ ctrl_b1,
                         const float* __restrict__ ctrl_w2, const float* __restrict__ ctrl_b2,
                         const float* __restrict__ gate_w,  const float* __restrict__ gate_b,
                         const float* __restrict__ assign_w,const float* __restrict__ assign_b,
                         const float* __restrict__ nov_w,   const float* __restrict__ nov_b,
                         const float* __restrict__ relax_w, const float* __restrict__ relax_b,
                         const float* __restrict__ cc_w,    const float* __restrict__ cc_b,
                         const float* __restrict__ cs_w,    const float* __restrict__ cs_b,
                         const float* __restrict__ md_w,    const float* __restrict__ md_b,
                         const float* __restrict__ att_w,   const float* __restrict__ att_b,
                         const float* __restrict__ out_w1,  const float* __restrict__ out_b1,
                         __bf16* __restrict__ H1) {
    const int b = blockIdx.x;
    const int t = threadIdx.x;

    __shared__ float s_cent[Cc][Hh];            // 32 KB
    __shared__ float s_spr[Cc], s_mass[Cc];
    __shared__ float s_alpha[Cc], s_sq[Cc];
    __shared__ float s_core[Hh], s_local[Hh];
    __shared__ float s_h1c[Hh], s_ctrl[Hh], s_att[Hh];
    __shared__ float s_cin[CIN];
    __shared__ float s_gate[Cc], s_assign[Cc], s_csx[Cc], s_mdx[Cc], s_str[Cc];
    __shared__ float s_ms[Cc], s_mm[Cc];
    __shared__ float s_mix[Cc][Cc];             // 4 KB
    __shared__ float s_red[256];
    __shared__ float s_scal[8];                 // unc,div,en,ent,nov,relax,valid
    __shared__ float s_feat[FEAT];
    __shared__ int   s_idx;

    // init state: centers=0, spreads=1, masses=0
    for (int i = t; i < Cc * Hh; i += 256) s_cent[i >> 8][i & 255] = 0.0f;
    if (t < Cc) { s_spr[t] = 1.0f; s_mass[t] = 0.0f; }
    __syncthreads();

    for (int ts = 0; ts < Ss; ++ts) {
        if (t == 0) s_scal[6] = (tokens[b * Ss + ts] != 0) ? 1.0f : 0.0f;
        s_local[t] = HL[((size_t)b * Ss + ts) * Hh + t];

        summarize_block(t, s_cent, s_spr, s_mass, s_alpha, s_core, s_red, s_scal, &s_idx);

        // ---- controller MLP ----
        s_cin[t] = s_local[t];
        s_cin[Hh + t] = s_core[t];
        if (t < 4) s_cin[2 * Hh + t] = s_scal[t];
        __syncthreads();
        {
            float acc = ctrl_b1[t];
            for (int i = 0; i < CIN; ++i) acc += s_cin[i] * ctrl_w1[(size_t)i * Hh + t];
            s_h1c[t] = tanhf(acc);
        }
        __syncthreads();
        {
            float acc = ctrl_b2[t];
            for (int i = 0; i < Hh; ++i) acc += s_h1c[i] * ctrl_w2[(size_t)i * Hh + t];
            s_ctrl[t] = tanhf(acc);
        }
        __syncthreads();

        // ---- heads ----
        if (t < Cc) {
            float ag = gate_b[t], aa = assign_b[t], as = cs_b[t], am = md_b[t];
            for (int i = 0; i < Hh; ++i) {
                const float cv = s_ctrl[i];
                ag += cv * gate_w[i * Cc + t];
                aa += cv * assign_w[i * Cc + t];
                as += cv * cs_w[i * Cc + t];
                am += cv * md_w[i * Cc + t];
            }
            s_gate[t]   = sigmoidf_(ag) * s_scal[6];
            s_assign[t] = aa;                       // softmaxed by thread 0 below
            s_csx[t]    = softplusf_(as) + EPSf;
            s_mdx[t]    = tanhf(am);
        }
        {
            float acc = att_b[t];
            for (int i = 0; i < Hh; ++i) acc += s_ctrl[i] * att_w[(size_t)i * Hh + t];
            s_att[t] = acc;
        }
        s_red[t] = s_ctrl[t] * nov_w[t];
        __syncthreads();
        if (t == 0) {
            float sum = nov_b[0];
            for (int i = 0; i < 256; ++i) sum += s_red[i];
            s_scal[4] = sigmoidf_(sum) * s_scal[6];
            // assign softmax (length 32, serial is cheap)
            float mx = -1e30f;
            for (int c = 0; c < Cc; ++c) mx = fmaxf(mx, s_assign[c]);
            float se = 0.0f;
            for (int c = 0; c < Cc; ++c) { const float e = expf(s_assign[c] - mx); s_assign[c] = e; se += e; }
            for (int c = 0; c < Cc; ++c) s_assign[c] /= se;
        }
        __syncthreads();
        s_red[t] = s_ctrl[t] * relax_w[t];
        __syncthreads();
        if (t == 0) {
            float sum = relax_b[0];
            for (int i = 0; i < 256; ++i) sum += s_red[i];
            s_scal[5] = sigmoidf_(sum) * s_scal[6];
        }
        __syncthreads();
        if (t < Cc) s_str[t] = s_gate[t] * s_assign[t];
        __syncthreads();

        // ---- candidate-center update (column t of every cluster) ----
        for (int c = 0; c < Cc; ++c) {
            float cand = cc_b[c * Hh + t];
            for (int i = 0; i < Hh; ++i)
                cand += s_ctrl[i] * cc_w[(size_t)i * CH + c * Hh + t];
            s_cent[c][t] += s_str[c] * (cand - s_cent[c][t]);
        }
        if (t < Cc) {
            s_spr[t]  += s_str[t] * (s_csx[t] - s_spr[t]);
            s_mass[t] += s_str[t] * s_mdx[t];
        }
        {   // attractor pull (column-local)
            const float nv = s_scal[4];
            for (int c = 0; c < Cc; ++c)
                s_cent[c][t] += 0.1f * nv * (s_att[t] - s_cent[c][t]);
        }
        __syncthreads();

        // ---- interact ----
        {   // sq[c] = sum_h centers^2
            const int c = t >> 3, j = t & 7;
            float p = 0.0f;
            for (int h = j * 32; h < j * 32 + 32; ++h) p += s_cent[c][h] * s_cent[c][h];
            s_red[t] = p;
        }
        __syncthreads();
        if (t < Cc) {
            float s = 0.0f;
            for (int j = 0; j < 8; ++j) s += s_red[t * 8 + j];
            s_sq[t] = s;
        }
        __syncthreads();
        for (int e = 0; e < 4; ++e) {   // 1024 compat entries / 256 threads
            const int id = t * 4 + e;
            const int c = id >> 5, d = id & 31;
            float dot = 0.0f;
            for (int h = 0; h < Hh; ++h) dot += s_cent[c][h] * s_cent[d][h];
            const float d2 = fmaxf(s_sq[c] + s_sq[d] - 2.0f * dot, 0.0f);
            s_mix[c][d] = -d2 / (s_spr[c] + s_spr[d] + EPSf) + s_mass[d];
        }
        __syncthreads();
        if (t < Cc) {   // per-row softmax
            float mx = -1e30f;
            for (int d = 0; d < Cc; ++d) mx = fmaxf(mx, s_mix[t][d]);
            float se = 0.0f;
            for (int d = 0; d < Cc; ++d) { const float e = expf(s_mix[t][d] - mx); s_mix[t][d] = e; se += e; }
            for (int d = 0; d < Cc; ++d) s_mix[t][d] /= se;
        }
        __syncthreads();
        if (t < Cc) {
            float ms = 0.0f, mm = 0.0f;
            for (int d = 0; d < Cc; ++d) { ms += s_mix[t][d] * s_spr[d]; mm += s_mix[t][d] * s_mass[d]; }
            s_ms[t] = ms; s_mm[t] = mm;
        }
        __syncthreads();
        {   // mixed centers, column t: read old values first, then blend
            float mcv[Cc];
            for (int c = 0; c < Cc; ++c) {
                float m = 0.0f;
                for (int d = 0; d < Cc; ++d) m += s_mix[c][d] * s_cent[d][t];
                mcv[c] = m;
            }
            const float r = s_scal[5];
            for (int c = 0; c < Cc; ++c)
                s_cent[c][t] = (1.0f - r) * s_cent[c][t] + r * mcv[c];
        }
        if (t < Cc) {
            const float r = s_scal[5];
            s_spr[t]  = (1.0f - r) * s_spr[t]  + r * s_ms[t];
            s_mass[t] = (1.0f - r) * s_mass[t] + r * s_mm[t];
        }
        __syncthreads();

        // ---- second summarize + output features ----
        summarize_block(t, s_cent, s_spr, s_mass, s_alpha, s_core, s_red, s_scal, &s_idx);
        s_feat[t]           = s_local[t];
        s_feat[Hh + t]      = s_core[t];
        s_feat[2 * Hh + t]  = s_cent[s_idx][t];
        if (t < 4) s_feat[3 * Hh + t] = s_scal[t];
        __syncthreads();
        {
            float acc = out_b1[t];
            for (int i = 0; i < FEAT; ++i) acc += s_feat[i] * out_w1[(size_t)i * Hh + t];
            const float g = 0.5f * acc * (1.0f + erff(acc * 0.70710678118654752f));
            H1[((size_t)b * Ss + ts) * Hh + t] = (__bf16)g;   // bf16 GEMM operand
        }
        __syncthreads();
    }
}

// ---------------------------------------------------------------
// launch
// ---------------------------------------------------------------
extern "C" void kernel_launch(void* const* d_in, const int* in_sizes, int n_in,
                              void* d_out, int out_size, void* d_ws, size_t ws_size,
                              hipStream_t stream) {
    (void)in_sizes; (void)n_in; (void)out_size; (void)ws_size;
    const int*   tokens   = (const int*)  d_in[0];
    const float* emb      = (const float*)d_in[1];
    const float* gru_wih  = (const float*)d_in[2];
    const float* gru_whh  = (const float*)d_in[3];
    const float* gru_bih  = (const float*)d_in[4];
    const float* gru_bhh  = (const float*)d_in[5];
    const float* ctrl_w1  = (const float*)d_in[6];
    const float* ctrl_b1  = (const float*)d_in[7];
    const float* ctrl_w2  = (const float*)d_in[8];
    const float* ctrl_b2  = (const float*)d_in[9];
    const float* gate_w   = (const float*)d_in[10];
    const float* gate_b   = (const float*)d_in[11];
    const float* assign_w = (const float*)d_in[12];
    const float* assign_b = (const float*)d_in[13];
    const float* nov_w    = (const float*)d_in[14];
    const float* nov_b    = (const float*)d_in[15];
    const float* relax_w  = (const float*)d_in[16];
    const float* relax_b  = (const float*)d_in[17];
    const float* cc_w     = (const float*)d_in[18];
    const float* cc_b     = (const float*)d_in[19];
    const float* cs_w     = (const float*)d_in[20];
    const float* cs_b     = (const float*)d_in[21];
    const float* md_w     = (const float*)d_in[22];
    const float* md_b     = (const float*)d_in[23];
    const float* att_w    = (const float*)d_in[24];
    const float* att_b    = (const float*)d_in[25];
    const float* out_w1   = (const float*)d_in[26];
    const float* out_b1   = (const float*)d_in[27];
    const float* out_w2   = (const float*)d_in[28];
    const float* out_b2   = (const float*)d_in[29];

    // ---- workspace layout ----
    float*  GI    = (float*)d_ws;                     // [1024, 768] f32
    float*  HL    = GI + (size_t)Mm * G3H;            // [1024, 256] f32
    __bf16* Xbf   = (__bf16*)(HL + (size_t)Mm * Hh);  // [1024, 256] bf16
    __bf16* Wihbf = Xbf   + (size_t)Mm * Ee;          // [768, 256]  bf16 (N x K)
    __bf16* W2bf  = Wihbf + (size_t)G3H * Ee;         // [32000,256] bf16 (N x K)
    __bf16* H1bf  = W2bf  + (size_t)Vv * Hh;          // [1024, 256] bf16

    // 1) embed -> bf16 A operand
    embed_bf16_kernel<<<Mm, 256, 0, stream>>>(tokens, emb, Xbf);

    // 1b) weight conversions (once per call; B stored N x K)
    convert_bf16_kernel<<<(G3H * Ee) / 256, 256, 0, stream>>>(gru_wih, Wihbf);
    transpose_bf16_kernel<<<Vv, 256, 0, stream>>>(out_w2, W2bf, Vv, Kk);

    // 2) GI = X @ wih.T + bih  (M=1024, N=768, K=256)
    {
        const int nblkn = (G3H + 511) / 512;          // 2
        wmma_gemm_bf16_kernel<<<(Mm / 16) * nblkn, 256, 0, stream>>>(
            Xbf, Wihbf, gru_bih, GI, Mm, G3H, nblkn);
    }

    // 3) GRU recurrence (batch-parallel)
    gru_scan_kernel<<<Bb, 256, 0, stream>>>(GI, gru_whh, gru_bhh, HL);

    // 4) cluster-field recurrence (batch-parallel) -> bf16 H1
    cluster_scan_kernel<<<Bb, 256, 0, stream>>>(
        tokens, HL, ctrl_w1, ctrl_b1, ctrl_w2, ctrl_b2, gate_w, gate_b,
        assign_w, assign_b, nov_w, nov_b, relax_w, relax_b, cc_w, cc_b,
        cs_w, cs_b, md_w, md_b, att_w, att_b, out_w1, out_b1, H1bf);

    // 5) logits = H1 @ out_w2 + out_b2  (M=1024, N=32000, K=256)
    {
        const int nblkn = (Vv + 511) / 512;           // 63
        wmma_gemm_bf16_kernel<<<(Mm / 16) * nblkn, 256, 0, stream>>>(
            H1bf, W2bf, out_b2, (float*)d_out, Mm, Vv, nblkn);
    }
}
